// TaxoHierMiner_51505247814435
// MI455X (gfx1250) — matrix operand order repaired and checked by reading, not compiled, not run on hardware
//
#include <hip/hip_runtime.h>
#include <hip/hip_bf16.h>

#define NEGV -1.0e9f

typedef __attribute__((ext_vector_type(16))) __bf16 v16bf;
typedef __attribute__((ext_vector_type(8)))  float  v8f;

union FragBF { v16bf v; uint4 q[2]; };

__device__ __forceinline__ unsigned short f2bf(float f) {
  unsigned int u = __float_as_uint(f);
  u += 0x7FFFu + ((u >> 16) & 1u);          // round-to-nearest-even
  return (unsigned short)(u >> 16);
}

// ---- CDNA5 async global->LDS copy (ASYNCcnt path), 16B per lane ----
__device__ __forceinline__ void async_cp16(unsigned int lds_off, const void* gptr) {
  asm volatile("global_load_async_to_lds_b128 %0, %1, off"
               :: "v"(lds_off), "v"((unsigned long long)(size_t)gptr)
               : "memory");
}
__device__ __forceinline__ void wait_async0() {
  asm volatile("s_wait_asynccnt 0" ::: "memory");
}

// thread t copies 2 of the 512 16B-chunks of a 128x32 bf16 tile into padded LDS
__device__ __forceinline__ void stage_tile_async(
    unsigned ldsBase, const unsigned short* __restrict__ gRowBase, int kk, int t)
{
  #pragma unroll
  for (int j = 0; j < 2; ++j) {
    int c = t * 2 + j, row = c >> 2, seg = c & 3;
    async_cp16(ldsBase + row * 80 + seg * 16,
               gRowBase + (size_t)row * 768 + kk + seg * 8);
  }
}

// -----------------------------------------------------------------------------
// Prepass A: f32 -> bf16 streaming convert (8 elems / thread)
// -----------------------------------------------------------------------------
__global__ __launch_bounds__(256) void k_cvt_bf16(
    const float* __restrict__ in, unsigned short* __restrict__ out, int n8)
{
  int i = blockIdx.x * 256 + threadIdx.x;
  if (i < n8) {
    const float4* p = (const float4*)(in + (size_t)i * 8);
    float4 f0 = p[0], f1 = p[1];
    union { uint4 q; unsigned short s[8]; } u;
    u.s[0] = f2bf(f0.x); u.s[1] = f2bf(f0.y); u.s[2] = f2bf(f0.z); u.s[3] = f2bf(f0.w);
    u.s[4] = f2bf(f1.x); u.s[5] = f2bf(f1.y); u.s[6] = f2bf(f1.z); u.s[7] = f2bf(f1.w);
    *(uint4*)(out + (size_t)i * 8) = u.q;
    if (i + 2048 < n8) __builtin_prefetch(in + ((size_t)i + 2048) * 8, 0, 1);
  }
}

// -----------------------------------------------------------------------------
// Prepass B: W[768,256] -> Wt[2][128][768] bf16, N-major (K contiguous)
// -----------------------------------------------------------------------------
__global__ __launch_bounds__(256) void k_prep_wt(
    const float* __restrict__ W, unsigned short* __restrict__ Wt)
{
  int idx = blockIdx.x * 256 + threadIdx.x;
  if (idx < 2 * 128 * 768) {
    int nb  = idx / (128 * 768);
    int rem = idx - nb * 128 * 768;
    int n   = rem / 768, k = rem - n * 768;
    Wt[idx] = f2bf(W[(size_t)k * 256 + nb * 128 + n]);
  }
}

// -----------------------------------------------------------------------------
// Kernel 1: reduce GEMM  out = X[M,768] @ W + bias   (bf16 WMMA, f32 acc)
// async global->LDS staging, ping-pong LDS, 8 back-to-back WMMAs per wave
// -----------------------------------------------------------------------------
__global__ __launch_bounds__(256) void k_reduce_gemm(
    const unsigned short* __restrict__ Xb,   // [M,768] bf16
    const unsigned short* __restrict__ Wt,   // [2*128,768] bf16 N-major
    const float* __restrict__ bias, float* __restrict__ outF,
    unsigned short* __restrict__ outB)
{
  __shared__ unsigned short As[2][128 * 40];  // padded rows: 64B data + 16B pad
  __shared__ unsigned short Bs[2][128 * 40];
  const int t    = threadIdx.x;
  const int lane = t & 31, wv = t >> 5;
  const int half = lane >> 4, n16 = lane & 15;
  const int mBase = blockIdx.x * 128;
  const int nBase = blockIdx.y * 128;
  const unsigned ldsA = (unsigned)(size_t)(void*)&As[0][0];
  const unsigned ldsB = (unsigned)(size_t)(void*)&Bs[0][0];
  const unsigned short* xRow = Xb + (size_t)mBase * 768;
  const unsigned short* wRow = Wt + (size_t)nBase * 768;

  v8f acc[8];
  #pragma unroll
  for (int i = 0; i < 8; ++i)
    #pragma unroll
    for (int j = 0; j < 8; ++j) acc[i][j] = 0.0f;

  // prologue: async-stage tile 0 into buffer 0
  stage_tile_async(ldsA, xRow, 0, t);
  stage_tile_async(ldsB, wRow, 0, t);
  wait_async0();
  __syncthreads();

  for (int kt = 0; kt < 24; ++kt) {          // 768 / 32
    const int buf = kt & 1;
    if (kt + 1 < 24) {                       // overlap next-tile fetch with WMMAs
      stage_tile_async(ldsA + (buf ^ 1) * 10240, xRow, (kt + 1) * 32, t);
      stage_tile_async(ldsB + (buf ^ 1) * 10240, wRow, (kt + 1) * 32, t);
    }
    // preload A frag + all 8 B frags, then issue WMMAs back-to-back
    FragBF a;
    {
      const unsigned short* abase = &As[buf][(wv * 16 + n16) * 40];
      a.q[0] = *(const uint4*)(abase + 16 * half);       // K = 16*half .. +7
      a.q[1] = *(const uint4*)(abase + 16 * half + 8);   // K = 16*half+8 .. +7
    }
    FragBF bfr[8];
    #pragma unroll
    for (int nt = 0; nt < 8; ++nt) {
      const unsigned short* bb = &Bs[buf][(nt * 16 + n16) * 40 + 16 * half];
      bfr[nt].q[0] = *(const uint4*)(bb);
      bfr[nt].q[1] = *(const uint4*)(bb + 8);
    }
    #pragma unroll
    for (int nt = 0; nt < 8; ++nt)
      acc[nt] = __builtin_amdgcn_wmma_f32_16x16x32_bf16(
          false, a.v, false, bfr[nt].v, (short)0, acc[nt], false, false);
    wait_async0();
    __syncthreads();
  }
  // epilogue: +bias, store f32 (+ optional bf16 copy)
  #pragma unroll
  for (int nt = 0; nt < 8; ++nt) {
    #pragma unroll
    for (int v = 0; v < 8; ++v) {
      int m   = mBase + wv * 16 + v + 8 * half;
      int col = nBase + nt * 16 + n16;
      float val = acc[nt][v] + bias[col];
      size_t o = (size_t)m * 256 + col;
      outF[o] = val;
      if (outB) outB[o] = f2bf(val);
    }
  }
}

// -----------------------------------------------------------------------------
// Kernel 2: per-row order-preserving dedup of subtopic ids (285-bit LDS bitset)
// -----------------------------------------------------------------------------
__global__ __launch_bounds__(256) void k_dedup_sub(
    const int* __restrict__ hsub, const unsigned char* __restrict__ hmask,
    int* __restrict__ sub_pad, int* __restrict__ sub_cnt)
{
  __shared__ unsigned int seen[256 * 9];
  int b = threadIdx.x;
  #pragma unroll
  for (int i = 0; i < 9; ++i) seen[b * 9 + i] = 0u;
  int cnt = 0;
  for (int l = 0; l < 200; ++l) {
    if (hmask[b * 200 + l]) {
      int id = hsub[b * 200 + l];
      unsigned int w = (unsigned)id >> 5, bit = 1u << (id & 31);
      if (!(seen[b * 9 + w] & bit)) {
        seen[b * 9 + w] |= bit;
        sub_pad[b * 200 + cnt] = id;
        ++cnt;
      }
    }
  }
  sub_cnt[b] = cnt;
}

// -----------------------------------------------------------------------------
// Kernel 3: subtopic attention pooling (fused)
//   scores (bf16 WMMA, D-contraction) -> masked softmax -> u_sub + topic vote
// grid = (13 k-tiles, B); block = 256
// -----------------------------------------------------------------------------
__global__ __launch_bounds__(256) void k_subpool(
    const float* __restrict__ q_sub_emb,
    const float* __restrict__ h_hist_f, const unsigned short* __restrict__ h_hist_b,
    const int* __restrict__ hsub, const int* __restrict__ htop,
    const unsigned char* __restrict__ hmask,
    const int* __restrict__ sub_pad, const int* __restrict__ sub_cnt,
    float* __restrict__ u_sub, int* __restrict__ topic_of)
{
  __shared__ unsigned short qa[16 * 264];   // 16 query rows x 256 bf16 (stride 264)
  __shared__ float sc[16 * 208];            // scores -> attention weights
  __shared__ int rowSid[16];
  const int kt = blockIdx.x;                // 0..12 (13 * 16 = 208 >= 200)
  const int b  = blockIdx.y;
  const int t  = threadIdx.x;
  const int lane = t & 31, wv = t >> 5;
  const int half = lane >> 4, n16 = lane & 15;
  const int cnt = sub_cnt[b];

  // ---- phase 0: gather + convert query rows ----
  {
    int r = t >> 4, dseg = (t & 15) * 16;
    int kr = kt * 16 + r;
    int sid = (kr < cnt) ? sub_pad[b * 200 + kr] : 0;
    const float4* p = (const float4*)(q_sub_emb + (size_t)sid * 256 + dseg);
    unsigned short* dst = qa + r * 264 + dseg;
    #pragma unroll
    for (int v = 0; v < 4; ++v) {
      float4 f = p[v];
      dst[v*4+0] = f2bf(f.x); dst[v*4+1] = f2bf(f.y);
      dst[v*4+2] = f2bf(f.z); dst[v*4+3] = f2bf(f.w);
    }
    if (t < 16) {
      int kr2 = kt * 16 + t;
      rowSid[t] = (kr2 < cnt) ? sub_pad[b * 200 + kr2] : 0;
    }
  }
  __syncthreads();

  // ---- phase 1: scores[16 x 208] = qa @ h_hist^T  (WMMA bf16, 8 K-steps) ----
  FragBF afr[8];                                  // hoisted: reused by both tiles
  #pragma unroll
  for (int ks = 0; ks < 8; ++ks) {
    const unsigned short* abase = qa + n16 * 264 + ks * 32;
    afr[ks].q[0] = *(const uint4*)(abase + 16 * half);
    afr[ks].q[1] = *(const uint4*)(abase + 16 * half + 8);
  }
  for (int lt = wv; lt < 13; lt += 8) {
    v8f acc;
    #pragma unroll
    for (int j = 0; j < 8; ++j) acc[j] = 0.0f;
    int l = lt * 16 + n16;
    int lsafe = (l < 200) ? l : 199;
    const unsigned short* hrow = h_hist_b + ((size_t)b * 200 + lsafe) * 256 + 16 * half;
    #pragma unroll
    for (int ks = 0; ks < 8; ++ks) {
      FragBF bb;
      const uint4* bp = (const uint4*)(hrow + ks * 32);
      bb.q[0] = bp[0];
      bb.q[1] = bp[1];
      acc = __builtin_amdgcn_wmma_f32_16x16x32_bf16(
          false, afr[ks].v, false, bb.v, (short)0, acc, false, false);
    }
    #pragma unroll
    for (int v = 0; v < 8; ++v)
      sc[(v + 8 * half) * 208 + lt * 16 + n16] = acc[v];
  }
  __syncthreads();

  // ---- phase 2: masked softmax per row + majority-vote topic ----
  for (int rr = 0; rr < 2; ++rr) {
    int r = wv * 2 + rr;
    int k = kt * 16 + r;
    if (k >= cnt) {                       // padded row: dead in the reference
      for (int l = lane; l < 208; l += 32) sc[r * 208 + l] = 0.0f;
      if (lane == 0 && k < 200) topic_of[b * 200 + k] = 0;
      continue;
    }
    int sid = rowSid[r];
    int c18[18];
    #pragma unroll
    for (int i = 0; i < 18; ++i) c18[i] = 0;
    float mx = -3.0e38f;
    for (int l = lane; l < 200; l += 32) {
      bool ok = (hmask[b * 200 + l] != 0) && (hsub[b * 200 + l] == sid);
      int tp = htop[b * 200 + l];
      float s = ok ? sc[r * 208 + l] : NEGV;
      sc[r * 208 + l] = s;
      mx = fmaxf(mx, s);
      #pragma unroll
      for (int q = 0; q < 18; ++q) c18[q] += (ok && tp == q) ? 1 : 0;
    }
    #pragma unroll
    for (int m = 16; m; m >>= 1) mx = fmaxf(mx, __shfl_xor(mx, m, 32));
    float sum = 0.f;
    for (int l = lane; l < 200; l += 32) {
      float e = __expf(sc[r * 208 + l] - mx);
      sc[r * 208 + l] = e;
      sum += e;
    }
    #pragma unroll
    for (int m = 16; m; m >>= 1) sum += __shfl_xor(sum, m, 32);
    float inv = 1.0f / sum;
    for (int l = lane; l < 200; l += 32) sc[r * 208 + l] *= inv;
    for (int l = 200 + lane; l < 208; l += 32) sc[r * 208 + l] = 0.0f;
    #pragma unroll
    for (int q = 0; q < 18; ++q) {
      int c = c18[q];
      #pragma unroll
      for (int m = 16; m; m >>= 1) c += __shfl_xor(c, m, 32);
      c18[q] = c;
    }
    if (lane == 0) {
      int best = 0, bc = c18[0];
      #pragma unroll
      for (int q = 1; q < 18; ++q) if (c18[q] > bc) { bc = c18[q]; best = q; }
      topic_of[b * 200 + k] = best;
    }
  }
  __syncthreads();

  // ---- phase 3: u_sub = attn @ h_hist (sparse: skip exact-zero weights) ----
  {
    int r = t >> 4, dseg = (t & 15) * 16;
    int k = kt * 16 + r;
    if (k < cnt) {
      float accv[16];
      #pragma unroll
      for (int i = 0; i < 16; ++i) accv[i] = 0.f;
      for (int l = 0; l < 200; ++l) {
        float a = sc[r * 208 + l];
        if (a != 0.0f) {
          const float4* hp = (const float4*)(h_hist_f + ((size_t)b * 200 + l) * 256 + dseg);
          #pragma unroll
          for (int v = 0; v < 4; ++v) {
            float4 f = hp[v];
            accv[v*4+0] += a * f.x; accv[v*4+1] += a * f.y;
            accv[v*4+2] += a * f.z; accv[v*4+3] += a * f.w;
          }
        }
      }
      float* dst = u_sub + ((size_t)b * 200 + k) * 256 + dseg;
      #pragma unroll
      for (int i = 0; i < 16; ++i) dst[i] = accv[i];
    }
  }
}

// -----------------------------------------------------------------------------
// Kernel 4: dedup topics of valid subtopics (<=18 -> one 32-bit bitset)
// -----------------------------------------------------------------------------
__global__ __launch_bounds__(256) void k_dedup_top(
    const int* __restrict__ topic_of, const int* __restrict__ sub_cnt,
    int* __restrict__ top_pad, int* __restrict__ top_cnt)
{
  int b = threadIdx.x;
  int cnt = sub_cnt[b];
  unsigned int seen = 0;
  int n = 0;
  for (int k = 0; k < cnt; ++k) {
    int id = topic_of[b * 200 + k];
    if (!((seen >> id) & 1u)) { seen |= (1u << id); top_pad[b * 32 + n] = id; ++n; }
  }
  top_cnt[b] = n;
}

// -----------------------------------------------------------------------------
// Kernel 5: topic pooling + normalize + disagreement partials (1 block / batch)
// -----------------------------------------------------------------------------
__global__ __launch_bounds__(256) void k_toppool(
    const float* __restrict__ q_top_emb, const float* __restrict__ u_sub,
    const int* __restrict__ sub_cnt, const int* __restrict__ topic_of,
    const int* __restrict__ top_pad, const int* __restrict__ top_cnt,
    float* __restrict__ u_top, float* __restrict__ u_topn,
    float* __restrict__ numArr, float* __restrict__ denArr)
{
  __shared__ float sbuf[208];
  __shared__ float red[256];
  __shared__ float un[18 * 256];
  const int b  = blockIdx.x;
  const int t0 = threadIdx.x;
  const int cnt = sub_cnt[b];
  const int Tn  = top_cnt[b];
  for (int ti = 0; ti < Tn; ++ti) {
    int tid = top_pad[b * 32 + ti];
    if (t0 < 208) {                           // thread t0 = subtopic index s
      float s = NEGV;
      if (t0 < cnt && topic_of[b * 200 + t0] == tid) {
        const float* q = q_top_emb + (size_t)tid * 256;
        const float* u = u_sub + ((size_t)b * 200 + t0) * 256;
        float d = 0.f;
        for (int i = 0; i < 256; i += 4)
          d += q[i]*u[i] + q[i+1]*u[i+1] + q[i+2]*u[i+2] + q[i+3]*u[i+3];
        s = d;
      }
      sbuf[t0] = s;
    }
    __syncthreads();
    red[t0] = (t0 < 208) ? sbuf[t0] : NEGV;
    __syncthreads();
    for (int st = 128; st > 0; st >>= 1) {
      if (t0 < st) red[t0] = fmaxf(red[t0], red[t0 + st]);
      __syncthreads();
    }
    float smax = red[0];
    __syncthreads();
    float e = (t0 < 208) ? __expf(sbuf[t0] - smax) : 0.0f;
    red[t0] = e;
    __syncthreads();
    for (int st = 128; st > 0; st >>= 1) {
      if (t0 < st) red[t0] += red[t0 + st];
      __syncthreads();
    }
    float ssum = red[0];
    __syncthreads();
    if (t0 < 208) sbuf[t0] = e / ssum;
    __syncthreads();
    // u_top[ti][d] with d = t0 (D == 256 == blockDim)
    float acc = 0.f;
    for (int s = 0; s < cnt; ++s) {
      float a = sbuf[s];
      if (a != 0.0f) acc += a * u_sub[((size_t)b * 200 + s) * 256 + t0];
    }
    red[t0] = acc * acc;
    __syncthreads();
    for (int st = 128; st > 0; st >>= 1) {
      if (t0 < st) red[t0] += red[t0 + st];
      __syncthreads();
    }
    float invn = 1.0f / fmaxf(sqrtf(red[0]), 1e-12f);
    __syncthreads();
    u_top[((size_t)b * 32 + ti) * 256 + t0] = acc;
    float nv = acc * invn;
    u_topn[((size_t)b * 32 + ti) * 256 + t0] = nv;
    un[ti * 256 + t0] = nv;
    __syncthreads();
  }
  // disagreement partials: sum_{i!=j} relu(n_i . n_j - 0.2); denom = Tn^2
  float numLocal = 0.f;
  for (int p = t0; p < Tn * Tn; p += 256) {
    int i = p / Tn, j = p - i * Tn;
    if (i != j) {
      const float* a = un + i * 256;
      const float* c = un + j * 256;
      float d = 0.f;
      for (int q = 0; q < 256; ++q) d += a[q] * c[q];
      float r = d - 0.2f;
      if (r > 0.f) numLocal += r;
    }
  }
  red[t0] = numLocal;
  __syncthreads();
  for (int st = 128; st > 0; st >>= 1) {
    if (t0 < st) red[t0] += red[t0 + st];
    __syncthreads();
  }
  if (t0 == 0) { numArr[b] = red[0]; denArr[b] = (float)(Tn * Tn); }
}

// -----------------------------------------------------------------------------
// Kernel 6: candidate-aware selection; y = sum_t gamma_t * (u_top[t] . h_cand)
// -----------------------------------------------------------------------------
__global__ __launch_bounds__(256) void k_candidate(
    const float* __restrict__ h_cand, const float* __restrict__ u_top,
    const float* __restrict__ u_topn, const int* __restrict__ top_cnt,
    float* __restrict__ y_hat)
{
  __shared__ float sU[18 * 256];
  __shared__ float sN[18 * 256];
  __shared__ float sdn[8][20];
  __shared__ float sdu[8][20];
  const int b  = blockIdx.x;
  const int Tn = top_cnt[b];
  for (int i = threadIdx.x; i < Tn * 256; i += 256) {
    sU[i] = u_top[(size_t)b * 32 * 256 + i];
    sN[i] = u_topn[(size_t)b * 32 * 256 + i];
  }
  __syncthreads();
  const int lane = threadIdx.x & 31, wv = threadIdx.x >> 5;
  for (int c = wv; c < 50; c += 8) {
    const float* hc = h_cand + ((size_t)b * 50 + c) * 256;
    const int d0 = lane * 8;
    float h[8];
    #pragma unroll
    for (int i = 0; i < 8; ++i) h[i] = hc[d0 + i];
    float n2 = 0.f;
    #pragma unroll
    for (int i = 0; i < 8; ++i) n2 += h[i] * h[i];
    #pragma unroll
    for (int m = 16; m; m >>= 1) n2 += __shfl_xor(n2, m, 32);
    const float invh = 1.0f / fmaxf(sqrtf(n2), 1e-12f);
    for (int ti = 0; ti < Tn; ++ti) {
      float dn = 0.f, du = 0.f;
      #pragma unroll
      for (int i = 0; i < 8; ++i) {
        dn += sN[ti * 256 + d0 + i] * h[i];
        du += sU[ti * 256 + d0 + i] * h[i];
      }
      #pragma unroll
      for (int m = 16; m; m >>= 1) { dn += __shfl_xor(dn, m, 32); du += __shfl_xor(du, m, 32); }
      if (lane == 0) { sdn[wv][ti] = dn * invh; sdu[wv][ti] = du; }
    }
    if (lane == 0) {
      float mx = -3.0e38f;
      for (int ti = 0; ti < Tn; ++ti) mx = fmaxf(mx, sdn[wv][ti]);
      float sum = 0.f, y = 0.f;
      for (int ti = 0; ti < Tn; ++ti) {
        float e = __expf(sdn[wv][ti] - mx);
        sum += e;
        y += e * sdu[wv][ti];
      }
      y_hat[b * 50 + c] = y / sum;
    }
  }
}

// -----------------------------------------------------------------------------
// Kernel 7: deterministic serial finalize of the disagreement loss
// -----------------------------------------------------------------------------
__global__ void k_finalize(const float* __restrict__ numArr,
                           const float* __restrict__ denArr,
                           float* __restrict__ out)
{
  if (threadIdx.x == 0 && blockIdx.x == 0) {
    float n = 0.f, d = 0.f;
    for (int b = 0; b < 256; ++b) { n += numArr[b]; d += denArr[b]; }
    out[256 * 50] = n / fmaxf(d, 1.0f);
  }
}

// -----------------------------------------------------------------------------
extern "C" void kernel_launch(void* const* d_in, const int* in_sizes, int n_in,
                              void* d_out, int out_size, void* d_ws, size_t ws_size,
                              hipStream_t stream) {
  (void)in_sizes; (void)n_in; (void)out_size; (void)ws_size;
  const float*         hist_cls = (const float*)d_in[0];   // [256,200,768]
  const float*         cand_cls = (const float*)d_in[1];   // [256,50,768]
  const float*         W        = (const float*)d_in[2];   // [768,256]
  const float*         bias     = (const float*)d_in[3];   // [256]
  const float*         q_sub    = (const float*)d_in[4];   // [285,256]
  const float*         q_top    = (const float*)d_in[5];   // [18,256]
  const int*           htop     = (const int*)d_in[6];     // [256,200]
  const int*           hsub     = (const int*)d_in[7];     // [256,200]
  const unsigned char* hmask    = (const unsigned char*)d_in[8]; // [256,200] bool
  float* out = (float*)d_out;                               // [256*50] y_hat ++ [1] dis

  char* w = (char*)d_ws;
  size_t off = 0;
  float*          h_hist_f = (float*)(w + off);          off += (size_t)256*200*256*4;
  unsigned short* h_hist_b = (unsigned short*)(w + off); off += (size_t)256*200*256*2;
  float*          h_cand_f = (float*)(w + off);          off += (size_t)256*50*256*4;
  float*          u_sub    = (float*)(w + off);          off += (size_t)256*200*256*4;
  unsigned short* xb_hist  = (unsigned short*)(w + off); off += (size_t)256*200*768*2;
  unsigned short* xb_cand  = (unsigned short*)(w + off); off += (size_t)256*50*768*2;
  unsigned short* wt_bf    = (unsigned short*)(w + off); off += (size_t)2*128*768*2;
  int*            sub_pad  = (int*)(w + off);            off += (size_t)256*200*4;
  int*            sub_cnt  = (int*)(w + off);            off += 1024;
  int*            topic_of = (int*)(w + off);            off += (size_t)256*200*4;
  int*            top_pad  = (int*)(w + off);            off += (size_t)256*32*4;
  int*            top_cnt  = (int*)(w + off);            off += 1024;
  float*          u_top    = (float*)(w + off);          off += (size_t)256*32*256*4;
  float*          u_topn   = (float*)(w + off);          off += (size_t)256*32*256*4;
  float*          numArr   = (float*)(w + off);          off += 1024;
  float*          denArr   = (float*)(w + off);          off += 1024;

  // prepasses: bf16 conversions + W transpose
  k_cvt_bf16<<<19200, 256, 0, stream>>>(hist_cls, xb_hist, 4915200);  // 39.3M elems
  k_cvt_bf16<<<4800, 256, 0, stream>>>(cand_cls, xb_cand, 1228800);   // 9.8M elems
  k_prep_wt<<<768, 256, 0, stream>>>(W, wt_bf);

  // main GEMMs (async-LDS staged, ping-pong, WMMA bf16)
  k_reduce_gemm<<<dim3(400, 2), 256, 0, stream>>>(xb_hist, wt_bf, bias, h_hist_f, h_hist_b);
  k_reduce_gemm<<<dim3(100, 2), 256, 0, stream>>>(xb_cand, wt_bf, bias, h_cand_f,
                                                  (unsigned short*)nullptr);

  k_dedup_sub<<<1, 256, 0, stream>>>(hsub, hmask, sub_pad, sub_cnt);
  k_subpool<<<dim3(13, 256), 256, 0, stream>>>(q_sub, h_hist_f, h_hist_b, hsub, htop, hmask,
                                               sub_pad, sub_cnt, u_sub, topic_of);
  k_dedup_top<<<1, 256, 0, stream>>>(topic_of, sub_cnt, top_pad, top_cnt);
  k_toppool<<<256, 256, 0, stream>>>(q_top, u_sub, sub_cnt, topic_of, top_pad, top_cnt,
                                     u_top, u_topn, numArr, denArr);
  k_candidate<<<256, 256, 0, stream>>>(h_cand_f, u_top, u_topn, top_cnt, out);
  k_finalize<<<1, 1, 0, stream>>>(numArr, denArr, out);
}
